// LSTM_shakespeare_47382079209824
// MI455X (gfx1250) — matrix operand order repaired and checked by reading, not compiled, not run on hardware
//
#include <hip/hip_runtime.h>

// LSTM shapes
#define B_TOT 1024
#define T_LEN 128
#define E_DIM 8
#define H_DIM 256
#define C_DIM 80
#define G4H   1024  // 4*H

typedef __attribute__((ext_vector_type(16))) __bf16    v16bf;
typedef __attribute__((ext_vector_type(8)))  float     v8f;
typedef __attribute__((ext_vector_type(4)))  unsigned  v4u;   // 16B carrier, builtin copy

typedef __attribute__((address_space(1))) const v4u            gv4u;
typedef __attribute__((address_space(1))) const unsigned short gushort;

union Frag32B { v16bf v; v4u q[2]; };

__device__ __forceinline__ unsigned short f2bf(float f) {
  unsigned int u = __float_as_uint(f);
  u += 0x7FFFu + ((u >> 16) & 1u);      // round-to-nearest-even
  return (unsigned short)(u >> 16);
}
__device__ __forceinline__ float bf2f(unsigned short s) {
  return __uint_as_float(((unsigned int)s) << 16);
}
__device__ __forceinline__ float sigm(float x)      { return 1.0f / (1.0f + __expf(-x)); }
__device__ __forceinline__ float tanh_fast(float x) { return 2.0f / (1.0f + __expf(-2.0f * x)) - 1.0f; }

__device__ __forceinline__ v8f wmma_bf16(v16bf a, v16bf b, v8f c) {
  // (neg_a, A, neg_b, B, c_mod, C, reuse_a, reuse_b)
  return __builtin_amdgcn_wmma_f32_16x16x32_bf16(false, a, false, b, (short)0, c, false, false);
}

// Optimization barrier on an addrspace(1) pointer: blocks LICM from hoisting
// the (loop-invariant) B-fragment loads out of the mp loop (round-2 showed a
// 512-VGPR alloc + scratch spills), while keeping loads in the GLOBAL address
// space -> global_load_b128 (LOADcnt only) instead of flat_load_b128.
__device__ __forceinline__ gushort* opaque_g(const unsigned short* p) {
  gushort* q = (gushort*)(unsigned long long)p;
  asm volatile("" : "+s"(q));
  return q;
}

__device__ __forceinline__ void load_bfrag(Frag32B& bb, gushort* base, int fragIdx, int lane) {
  gv4u* p = (gv4u*)(base + fragIdx * 512) + lane * 2;
  bb.q[0] = p[0]; bb.q[1] = p[1];
}
__device__ __forceinline__ void load_afrag(Frag32B& aa, const unsigned short* p) {
  aa.q[0] = *(const v4u*)(p); aa.q[1] = *(const v4u*)(p + 16);
}

// ---------------------------------------------------------------------------
// Pack fp32 weights into wave-fragment-major bf16 layout for WMMA B operands.
// B-tile (32x16) for (n_tile, k_step): lane L holds column n_tile*16+(L&15),
// K = (j>>3)*16 + (L>>4)*8 + (j&7) for j=0..15, stored as 32 contiguous bytes per lane.
// w_ih0 (K=8) is zero-padded to one K=32 step.
// ---------------------------------------------------------------------------
__global__ void pack_weights(const float* __restrict__ w_ih0, const float* __restrict__ w_hh0,
                             const float* __restrict__ w_ih1, const float* __restrict__ w_hh1,
                             const float* __restrict__ b_ih0, const float* __restrict__ b_hh0,
                             const float* __restrict__ b_ih1, const float* __restrict__ b_hh1,
                             unsigned short* __restrict__ ih0f, unsigned short* __restrict__ hh0f,
                             unsigned short* __restrict__ ih1f, unsigned short* __restrict__ hh1f,
                             float* __restrict__ b0, float* __restrict__ b1) {
  const int NIH0 = 64 * 512;    // 64 n-tiles * 512 vals
  const int NHH  = 512 * 512;   // 64 n-tiles * 8 k-steps * 512 vals
  const int stride = gridDim.x * blockDim.x;
  for (int idx = blockIdx.x * blockDim.x + threadIdx.x; idx < NIH0 + 3 * NHH; idx += stride) {
    if (idx < NIH0) {
      int n = idx >> 9, r = idx & 511, L = r >> 4, j = r & 15;
      int k = ((j >> 3) << 4) + ((L >> 4) << 3) + (j & 7);
      int g = (n << 4) + (L & 15);
      ih0f[idx] = (k < E_DIM) ? f2bf(w_ih0[g * E_DIM + k]) : (unsigned short)0;
    } else {
      int t = idx - NIH0;
      int seg = t / NHH;                      // 0:hh0 1:ih1 2:hh1
      int local = t - seg * NHH;
      int frag = local >> 9, r = local & 511, L = r >> 4, j = r & 15;
      int n = frag >> 3, kk = frag & 7;
      int k = kk * 32 + ((j >> 3) << 4) + ((L >> 4) << 3) + (j & 7);
      int g = (n << 4) + (L & 15);
      const float* src = (seg == 0) ? w_hh0 : (seg == 1) ? w_ih1 : w_hh1;
      unsigned short* dst = (seg == 0) ? hh0f : (seg == 1) ? ih1f : hh1f;
      dst[local] = f2bf(src[g * H_DIM + k]);
    }
  }
  for (int i = blockIdx.x * blockDim.x + threadIdx.x; i < G4H; i += stride) {
    b0[i] = b_ih0[i] + b_hh0[i];
    b1[i] = b_ih1[i] + b_hh1[i];
  }
}

// ---------------------------------------------------------------------------
// Persistent 2-layer LSTM. 16 WGs x 256 threads (8 waves). Each WG owns a
// 64-row batch tile, loops all 128 timesteps with h (bf16, ping-pong) and
// c (f32) resident in dynamic LDS. Wave w owns hidden cols [32w, 32w+32).
// GEMM passes are (sub, mp): one 16-col hidden sub-tile x all 4 gates x
// 2 M-tiles -> acc = 8 tiles = 64 VGPRs. A/B fragments are explicitly
// double-buffered (1-step software pipeline) to keep ~10 loads in flight
// under the 8 WMMAs of each k-step (round-5 asm showed single-buffered B).
// ---------------------------------------------------------------------------
__global__ __launch_bounds__(256, 1)
void lstm_persistent(const int* __restrict__ x, const float* __restrict__ emb,
                     const unsigned short* __restrict__ ih0f, const unsigned short* __restrict__ hh0f,
                     const unsigned short* __restrict__ ih1f, const unsigned short* __restrict__ hh1f,
                     const float* __restrict__ b0g, const float* __restrict__ b1g,
                     const float* __restrict__ w_out, const float* __restrict__ b_out,
                     float* __restrict__ out) {
  extern __shared__ char smem[];
  unsigned short* h0buf = (unsigned short*)smem;            // 2 x 64 x 256 bf16
  unsigned short* h1buf = h0buf + 2 * 64 * 256;             // 2 x 64 x 256 bf16
  float* c0 = (float*)(h1buf + 2 * 64 * 256);               // 64 x 256 f32
  float* c1 = c0 + 64 * 256;                                // 64 x 256 f32
  unsigned short* elds = (unsigned short*)(c1 + 64 * 256);  // 64 x 32 bf16 (K-padded)
  float* b0l = (float*)(elds + 64 * 32);                    // 1024 f32
  float* b1l = b0l + G4H;                                   // 1024 f32

  const int tid  = threadIdx.x;
  const int lane = tid & 31;
  const int w    = tid >> 5;          // wave id 0..7 -> hidden cols [32w,32w+32)
  const int g16  = lane >> 4;
  const int l16  = lane & 15;
  const int b_base = blockIdx.x * 64;
  const v8f vzero = {0.f, 0.f, 0.f, 0.f, 0.f, 0.f, 0.f, 0.f};

  for (int i = tid; i < 64 * 256; i += 256) {
    h0buf[i] = 0; h0buf[16384 + i] = 0;
    h1buf[i] = 0; h1buf[16384 + i] = 0;
    c0[i] = 0.0f; c1[i] = 0.0f;
  }
  for (int i = tid; i < 64 * 32; i += 256) elds[i] = 0;  // K pad stays 0 forever
  for (int i = tid; i < G4H; i += 256) { b0l[i] = b0g[i]; b1l[i] = b1g[i]; }
  __syncthreads();

  for (int t = 0; t < T_LEN; ++t) {
    // stage embedding rows for this timestep (2 values per thread)
    {
      int r = tid >> 2, e0 = (tid & 3) * 2;
      int ci = x[(b_base + r) * T_LEN + t];
      elds[r * 32 + e0]     = f2bf(emb[ci * E_DIM + e0]);
      elds[r * 32 + e0 + 1] = f2bf(emb[ci * E_DIM + e0 + 1]);
    }
    __syncthreads();
    const unsigned short* h0c = h0buf + ((t & 1) * 16384);
    unsigned short*       h0n = h0buf + (((t & 1) ^ 1) * 16384);
    const unsigned short* h1c = h1buf + ((t & 1) * 16384);
    unsigned short*       h1n = h1buf + (((t & 1) ^ 1) * 16384);

    // ------------------------- layer 0 -------------------------
    // K-stream: step 0 = embedding (zero-padded K=32, B from ih0f),
    //           steps 1..8 = h0(t-1) k-chunks (B from hh0f).
    #pragma unroll
    for (int sub = 0; sub < 2; ++sub) {
      #pragma unroll 1
      for (int mp = 0; mp < 2; ++mp) {        // 32-row M block, 2 M-tiles
        gushort* ih0p = opaque_g(ih0f);
        gushort* hh0p = opaque_g(hh0f);
        const int tb = 2 * w + sub;           // tile = g*16 + tb
        v8f acc[2][4];
        #pragma unroll
        for (int m2 = 0; m2 < 2; ++m2)
          #pragma unroll
          for (int g = 0; g < 4; ++g) acc[m2][g] = vzero;

        Frag32B bbuf[2][4], abuf[2][2];
        // preload step 0: B from ih0f, A from embedding LDS
        #pragma unroll
        for (int g = 0; g < 4; ++g) load_bfrag(bbuf[0][g], ih0p, g * 16 + tb, lane);
        {
          const unsigned short* ep = elds + (mp * 32 + l16) * 32 + g16 * 8;
          load_afrag(abuf[0][0], ep);
          load_afrag(abuf[0][1], ep + 16 * 32);
        }
        #pragma unroll
        for (int s = 0; s <= 8; ++s) {
          int cur = s & 1, nxt = cur ^ 1;
          if (s < 8) {  // prefetch step s+1: B from hh0f k-chunk s, A from h0c k-chunk s
            #pragma unroll
            for (int g = 0; g < 4; ++g)
              load_bfrag(bbuf[nxt][g], hh0p, (g * 16 + tb) * 8 + s, lane);
            const unsigned short* hp = h0c + (mp * 32 + l16) * 256 + s * 32 + g16 * 8;
            load_afrag(abuf[nxt][0], hp);
            load_afrag(abuf[nxt][1], hp + 16 * 256);
          }
          #pragma unroll
          for (int g = 0; g < 4; ++g) {
            acc[0][g] = wmma_bf16(abuf[cur][0].v, bbuf[cur][g].v, acc[0][g]);
            acc[1][g] = wmma_bf16(abuf[cur][1].v, bbuf[cur][g].v, acc[1][g]);
          }
        }
        // activation: i,f,g,o all present in this pass
        #pragma unroll
        for (int m2 = 0; m2 < 2; ++m2) {
          int col = 32 * w + sub * 16 + l16;
          float bi = b0l[col], bf = b0l[256 + col], bg = b0l[512 + col], bo = b0l[768 + col];
          #pragma unroll
          for (int r = 0; r < 8; ++r) {
            int row = mp * 32 + m2 * 16 + 8 * g16 + r;
            float iv = acc[m2][0][r] + bi;
            float fv = acc[m2][1][r] + bf;
            float gv = acc[m2][2][r] + bg;
            float ov = acc[m2][3][r] + bo;
            float co = c0[row * 256 + col];
            float cn = sigm(fv) * co + sigm(iv) * tanh_fast(gv);
            float hn = sigm(ov) * tanh_fast(cn);
            c0[row * 256 + col] = cn;
            h0n[row * 256 + col] = f2bf(hn);
          }
        }
      }
    }
    __syncthreads();

    // --------- layer 1 (fused: h0(t)@w_ih1^T + h1(t-1)@w_hh1^T) ---------
    // K-stream: steps 0..7 = h0(t) vs w_ih1; steps 8..15 = h1(t-1) vs w_hh1.
    #pragma unroll
    for (int sub = 0; sub < 2; ++sub) {
      #pragma unroll 1
      for (int mp = 0; mp < 2; ++mp) {
        gushort* ih1p = opaque_g(ih1f);
        gushort* hh1p = opaque_g(hh1f);
        const int tb = 2 * w + sub;
        v8f acc[2][4];
        #pragma unroll
        for (int m2 = 0; m2 < 2; ++m2)
          #pragma unroll
          for (int g = 0; g < 4; ++g) acc[m2][g] = vzero;

        Frag32B bbuf[2][4], abuf[2][2];
        // preload step 0: B from ih1f kk=0, A from h0n kk=0
        #pragma unroll
        for (int g = 0; g < 4; ++g) load_bfrag(bbuf[0][g], ih1p, (g * 16 + tb) * 8, lane);
        {
          const unsigned short* hp = h0n + (mp * 32 + l16) * 256 + g16 * 8;
          load_afrag(abuf[0][0], hp);
          load_afrag(abuf[0][1], hp + 16 * 256);
        }
        #pragma unroll
        for (int s = 0; s <= 15; ++s) {
          int cur = s & 1, nxt = cur ^ 1;
          if (s < 15) {  // prefetch step s+1
            int sn = s + 1;
            gushort* wsrc = (sn < 8) ? ih1p : hh1p;
            const unsigned short* hsrc = (sn < 8) ? h0n : h1c;
            int kks = sn & 7;
            #pragma unroll
            for (int g = 0; g < 4; ++g)
              load_bfrag(bbuf[nxt][g], wsrc, (g * 16 + tb) * 8 + kks, lane);
            const unsigned short* hp = hsrc + (mp * 32 + l16) * 256 + kks * 32 + g16 * 8;
            load_afrag(abuf[nxt][0], hp);
            load_afrag(abuf[nxt][1], hp + 16 * 256);
          }
          #pragma unroll
          for (int g = 0; g < 4; ++g) {
            acc[0][g] = wmma_bf16(abuf[cur][0].v, bbuf[cur][g].v, acc[0][g]);
            acc[1][g] = wmma_bf16(abuf[cur][1].v, bbuf[cur][g].v, acc[1][g]);
          }
        }
        #pragma unroll
        for (int m2 = 0; m2 < 2; ++m2) {
          int col = 32 * w + sub * 16 + l16;
          float bi = b1l[col], bf = b1l[256 + col], bg = b1l[512 + col], bo = b1l[768 + col];
          #pragma unroll
          for (int r = 0; r < 8; ++r) {
            int row = mp * 32 + m2 * 16 + 8 * g16 + r;
            float iv = acc[m2][0][r] + bi;
            float fv = acc[m2][1][r] + bf;
            float gv = acc[m2][2][r] + bg;
            float ov = acc[m2][3][r] + bo;
            float co = c1[row * 256 + col];
            float cn = sigm(fv) * co + sigm(iv) * tanh_fast(gv);
            float hn = sigm(ov) * tanh_fast(cn);
            c1[row * 256 + col] = cn;
            h1n[row * 256 + col] = f2bf(hn);
          }
        }
      }
    }
    __syncthreads();
  }

  // after t=127 (cur=1 wrote nxt=0): final h1 lives in buffer 0
  const unsigned short* hf = h1buf;
  // logits = h1 @ w_out^T + b_out   (tiny: 64x80, VALU dot products)
  for (int i = tid; i < 64 * C_DIM; i += 256) {
    int row = i / C_DIM, col = i - row * C_DIM;
    float s = b_out[col];
    const unsigned short* hr = hf + row * 256;
    const float* wr = w_out + col * 256;
    #pragma unroll 8
    for (int k = 0; k < 256; ++k) s += bf2f(hr[k]) * wr[k];
    out[(b_base + row) * C_DIM + col] = s;
  }
  // h output (f32)
  for (int i = tid; i < 64 * 256; i += 256) {
    int row = i >> 8, col = i & 255;
    out[B_TOT * C_DIM + (b_base + row) * 256 + col] = bf2f(hf[i]);
  }
}

extern "C" void kernel_launch(void* const* d_in, const int* in_sizes, int n_in,
                              void* d_out, int out_size, void* d_ws, size_t ws_size,
                              hipStream_t stream) {
  (void)in_sizes; (void)n_in; (void)out_size; (void)ws_size;
  const int*   x     = (const int*)  d_in[0];
  const float* emb   = (const float*)d_in[1];
  const float* w_ih0 = (const float*)d_in[2];
  const float* w_hh0 = (const float*)d_in[3];
  const float* b_ih0 = (const float*)d_in[4];
  const float* b_hh0 = (const float*)d_in[5];
  const float* w_ih1 = (const float*)d_in[6];
  const float* w_hh1 = (const float*)d_in[7];
  const float* b_ih1 = (const float*)d_in[8];
  const float* b_hh1 = (const float*)d_in[9];
  const float* w_out = (const float*)d_in[10];
  const float* b_out = (const float*)d_in[11];

  char* ws = (char*)d_ws;
  unsigned short* ih0f = (unsigned short*)(ws);
  unsigned short* hh0f = (unsigned short*)(ws + 65536);
  unsigned short* ih1f = (unsigned short*)(ws + 65536 + 524288);
  unsigned short* hh1f = (unsigned short*)(ws + 65536 + 2 * 524288);
  float* b0 = (float*)(ws + 65536 + 3 * 524288);
  float* b1 = (float*)(ws + 65536 + 3 * 524288 + 4096);

  pack_weights<<<dim3(512), dim3(256), 0, stream>>>(
      w_ih0, w_hh0, w_ih1, w_hh1, b_ih0, b_hh0, b_ih1, b_hh1,
      ih0f, hh0f, ih1f, hh1f, b0, b1);

  const size_t smem_bytes = 274432;  // 128K h (ping-pong) + 128K c + e/bias
  (void)hipFuncSetAttribute(reinterpret_cast<const void*>(lstm_persistent),
                            hipFuncAttributeMaxDynamicSharedMemorySize, (int)smem_bytes);
  lstm_persistent<<<dim3(B_TOT / 64), dim3(256), smem_bytes, stream>>>(
      x, emb, ih0f, hh0f, ih1f, hh1f, b0, b1, w_out, b_out, (float*)d_out);
}
